// Locoformer_37409165148673
// MI455X (gfx1250) — compile-verified, matrix operand order checked by reading, not run on hardware
//
#include <hip/hip_runtime.h>
#include <hip/hip_bf16.h>

// ---------------------------------------------------------------------------
// Locoformer (XL-attention) for MI455X / gfx1250, wave32 + WMMA bf16.
// All WMMA operands pre-converted to bf16 and laid out so every fragment
// load is a contiguous 16B (b128) access.
// ---------------------------------------------------------------------------

typedef __attribute__((ext_vector_type(16))) __bf16 v16bf;
typedef __attribute__((ext_vector_type(8)))  __bf16 v8bf;
typedef __attribute__((ext_vector_type(8)))  float  v8f;

#define WMMA_BF16(A, B, C) \
  __builtin_amdgcn_wmma_f32_16x16x32_bf16(false, (A), false, (B), (short)0, (C), false, false)

static __device__ __forceinline__ v16bf cat16(v8bf lo, v8bf hi) {
  return __builtin_shufflevector(lo, hi, 0, 1, 2, 3, 4, 5, 6, 7,
                                         8, 9, 10, 11, 12, 13, 14, 15);
}

static constexpr int Nq     = 2048;   // query tokens
static constexpr int Dm     = 512;    // model dim
static constexpr int Hh     = 8;      // heads
static constexpr int DHd    = 64;     // head dim
static constexpr int MEMLEN = 512;    // cached tokens
static constexpr int KVLEN  = 2560;   // MEM + N

// ---------------------------------------------------------------------------
// One-shot converters (bf16 happens once per element, never in hot loops).
// ---------------------------------------------------------------------------
__global__ void __launch_bounds__(256)
cvt_bf16(const float* __restrict__ in, __bf16* __restrict__ out, int n) {
  const int i = blockIdx.x * 256 + threadIdx.x;
  if (i < n) out[i] = (__bf16)in[i];
}

// fp32 [K,N] -> bf16 [N,K]  (weights transposed so B-fragment loads are
// contiguous along the contraction dim)
__global__ void __launch_bounds__(256)
cvt_bf16_t(const float* __restrict__ in, __bf16* __restrict__ out, int K, int N) {
  const int i = blockIdx.x * 256 + threadIdx.x;
  if (i >= K * N) return;
  const int k = i / N, n = i - k * N;
  out[(size_t)n * K + k] = (__bf16)in[i];
}

// ---------------------------------------------------------------------------
// GEMM: C[M,N] = A[M,K] * Bt[N,K]^T   (bf16 in, fp32 out, WMMA 16x16x32)
// One wave per 16x16 C tile. Every fragment load = 16B contiguous chunk.
// M%16==0, K%32==0; N arbitrary (loads clamped, stores guarded).
// ---------------------------------------------------------------------------
__global__ void __launch_bounds__(128)
gemm_bf16_wmma(const __bf16* __restrict__ A, const __bf16* __restrict__ Bt,
               float* __restrict__ C, int M, int K, int N) {
  const int wave = blockIdx.x * (blockDim.x >> 5) + (threadIdx.x >> 5);
  const int ntN = (N + 15) >> 4;
  const int ntM = M >> 4;
  if (wave >= ntM * ntN) return;                 // wave-uniform exit
  const int tm = wave / ntN, tn = wave - tm * ntN;
  const int lane = threadIdx.x & 31, g = lane >> 4, r = lane & 15;

  const int rowA  = tm * 16 + r;
  const int colB  = tn * 16 + r;
  const int colLd = (colB < N) ? colB : (N - 1); // clamp OOB load column

  v8f acc = {0.f, 0.f, 0.f, 0.f, 0.f, 0.f, 0.f, 0.f};
  const v8bf* apv = (const v8bf*)(A  + (size_t)rowA  * K);
  const v8bf* bpv = (const v8bf*)(Bt + (size_t)colLd * K);

  for (int k0 = 0; k0 < K; k0 += 32) {
    const int cb = k0 >> 3;                      // 8-element chunk index
    __builtin_prefetch(apv + cb + 8, 0, 1);      // global_prefetch_b8 (next tile)
    // A: lane = M row; K half-split per ISA layout  (chunks g, 2+g)
    const v16bf a = cat16(apv[cb + g], apv[cb + 2 + g]);
    // B: lane = N col; same K striping over Bt row
    const v16bf b = cat16(bpv[cb + g], bpv[cb + 2 + g]);
    acc = WMMA_BF16(a, b, acc);
  }

  if (colB < N) {
#pragma unroll
    for (int i = 0; i < 8; ++i)                  // C: vgpr i -> M = g*8+i
      C[(size_t)(tm * 16 + g * 8 + i) * N + colB] = acc[i];
  }
}

// ---------------------------------------------------------------------------
// RoPE (GPT-NeoX half-rotation) on Q + SCALE=DH^-0.5, fp32 in -> bf16 out.
// ---------------------------------------------------------------------------
__global__ void __launch_bounds__(256)
rope_scale_q(const float* __restrict__ Qp, __bf16* __restrict__ Qr) {
  const int tid = blockIdx.x * 256 + threadIdx.x;
  if (tid >= Nq * Dm) return;
  const int n = tid >> 9;                        // / Dm (512)
  const int c = tid & (Dm - 1);
  const int d = c & (DHd - 1);
  const int j = d & 31;
  const float pos = (float)(MEMLEN + n);
  const float fr  = pos * __powf(10000.0f, -(float)j * (1.0f / 32.0f));
  const float cs = __cosf(fr), sn = __sinf(fr);
  const float xv = Qp[tid];
  const float ov = (d < 32) ? Qp[tid + 32] : Qp[tid - 32];
  const float rt = (d < 32) ? -ov : ov;
  Qr[tid] = (__bf16)((xv * cs + rt * sn) * 0.125f);   // * DH^-0.5
}

// ---------------------------------------------------------------------------
// Build bf16 K_full[H,KV,DH] (RoPE at pos=kv) and bf16 V^T[H,DH,KV] from
// cache_k/cache_v [H,MEM,DH] (fp32) and the KV projection [N, 2*H*DH] (fp32).
// V is stored transposed so attention's B-fragment loads are contiguous.
// ---------------------------------------------------------------------------
__global__ void __launch_bounds__(256)
build_kv_rope(const float* __restrict__ ck, const float* __restrict__ cv,
              const float* __restrict__ KVp, __bf16* __restrict__ Kf,
              __bf16* __restrict__ Vt) {
  const int tid = blockIdx.x * 256 + threadIdx.x;
  if (tid >= Hh * KVLEN * DHd) return;
  const int d   = tid & (DHd - 1);
  const int kvi = (tid >> 6) % KVLEN;
  const int h   = tid / (KVLEN * DHd);
  const int dob = (d < 32) ? d + 32 : d - 32;
  const bool cached = (kvi < MEMLEN);
  const size_t cbase = ((size_t)h * MEMLEN + kvi) * DHd;
  const size_t pbase = (size_t)(kvi - MEMLEN) * (2 * Dm) + (size_t)h * DHd;

  const float vv = cached ? cv[cbase + d]   : KVp[pbase + Dm + d];
  const float kx = cached ? ck[cbase + d]   : KVp[pbase + d];
  const float ko = cached ? ck[cbase + dob] : KVp[pbase + dob];

  const float fr = (float)kvi * __powf(10000.0f, -(float)(d & 31) * (1.0f / 32.0f));
  const float cs = __cosf(fr), sn = __sinf(fr);
  const float rt = (d < 32) ? -ko : ko;

  Kf[tid] = (__bf16)(kx * cs + rt * sn);                         // [h][kv][d]
  Vt[((size_t)h * DHd + d) * KVLEN + kvi] = (__bf16)vv;          // [h][d][kv]
}

// ---------------------------------------------------------------------------
// Flash-style masked attention. One wave per (16-query tile, head).
// Block window + lookback -> key loop range [ (bq-1)*512 , q_pos_max ];
// causal + episode equality via VALU selects (EXEC stays full for WMMA).
// Scores: 2x wmma per 16-key subtile; PV: 4x wmma per 32 keys, with P
// re-laid-out C-frag -> A-frag through 1KB of LDS (b128 ds loads).
// Sigmoid gate fused in epilogue; output written bf16 for the Wo GEMM.
// ---------------------------------------------------------------------------
__global__ void __launch_bounds__(32)
attn_wmma(const __bf16* __restrict__ Qr, const __bf16* __restrict__ Kf,
          const __bf16* __restrict__ Vt, const int* __restrict__ eids,
          const float* __restrict__ Gp, __bf16* __restrict__ Out) {
  const int qt = blockIdx.x, h = blockIdx.y;
  const int q0 = qt * 16;
  const int lane = threadIdx.x & 31, g = lane >> 4, r = lane & 15;
  __shared__ __bf16 Plds[16 * 32];

  // Q A-fragments for dh 0..31 / 32..63 (lane = query row q0+r), b128 loads.
  const v8bf* qv = (const v8bf*)(Qr + (size_t)(q0 + r) * Dm + h * DHd);
  const v16bf qa0 = cat16(qv[g],     qv[2 + g]);
  const v16bf qa1 = cat16(qv[4 + g], qv[6 + g]);

  int eq[8];
#pragma unroll
  for (int i = 0; i < 8; ++i) eq[i] = eids[MEMLEN + q0 + g * 8 + i];

  float mrow[8], lrow[8];
#pragma unroll
  for (int i = 0; i < 8; ++i) { mrow[i] = -1e30f; lrow[i] = 0.f; }
  v8f acc[4];
#pragma unroll
  for (int c = 0; c < 4; ++c)
#pragma unroll
    for (int i = 0; i < 8; ++i) acc[c][i] = 0.f;

  const int lo = (q0 / 512) * 512;          // (bq-1)*WINDOW
  const int hi = MEMLEN + q0 + 15;          // max q_pos in tile (<= KV-1)

  for (int kt = lo; kt <= hi; kt += 32) {
    const int ek0 = eids[kt + r];
    const int ek1 = eids[kt + 16 + r];

    // ---- scores for two 16-key subtiles (K rows contiguous in d) ------
    v8f s0, s1;
    {
      const v8bf* kv0 = (const v8bf*)(Kf + ((size_t)h * KVLEN + (kt + r)) * DHd);
      __builtin_prefetch(kv0 + 256, 0, 1);   // next 32-key tile of K
      v8f t = {0.f,0.f,0.f,0.f,0.f,0.f,0.f,0.f};
      t  = WMMA_BF16(qa0, cat16(kv0[g],     kv0[2 + g]), t);
      s0 = WMMA_BF16(qa1, cat16(kv0[4 + g], kv0[6 + g]), t);
    }
    {
      const v8bf* kv1 = (const v8bf*)(Kf + ((size_t)h * KVLEN + (kt + 16 + r)) * DHd);
      v8f t = {0.f,0.f,0.f,0.f,0.f,0.f,0.f,0.f};
      t  = WMMA_BF16(qa0, cat16(kv1[g],     kv1[2 + g]), t);
      s1 = WMMA_BF16(qa1, cat16(kv1[4 + g], kv1[6 + g]), t);
    }

    // ---- mask + online softmax update (rows live in lane groups) ------
    float alph[8];
#pragma unroll
    for (int i = 0; i < 8; ++i) {
      const int qpos = MEMLEN + q0 + g * 8 + i;
      const int key0 = kt + r, key1 = kt + 16 + r;
      float v0 = ((key0 <= qpos) && (ek0 == eq[i])) ? s0[i] : -1e30f;
      float v1 = ((key1 <= qpos) && (ek1 == eq[i])) ? s1[i] : -1e30f;
      float mx = fmaxf(v0, v1);
#pragma unroll
      for (int msk = 1; msk < 16; msk <<= 1) mx = fmaxf(mx, __shfl_xor(mx, msk, 32));
      const float nm = fmaxf(mrow[i], mx);
      const float p0 = (v0 > -1e29f) ? __expf(v0 - nm) : 0.f;
      const float p1 = (v1 > -1e29f) ? __expf(v1 - nm) : 0.f;
      Plds[(g * 8 + i) * 32 + r]      = (__bf16)p0;
      Plds[(g * 8 + i) * 32 + 16 + r] = (__bf16)p1;
      float ps = p0 + p1;
#pragma unroll
      for (int msk = 1; msk < 16; msk <<= 1) ps += __shfl_xor(ps, msk, 32);
      const float al = __expf(mrow[i] - nm);
      lrow[i] = lrow[i] * al + ps;
      mrow[i] = nm;
      alph[i] = al;
    }
#pragma unroll
    for (int c = 0; c < 4; ++c)
#pragma unroll
      for (int i = 0; i < 8; ++i) acc[c][i] *= alph[i];

    __syncthreads();                       // P tile visible wave-wide

    // ---- P(16x32 keys) . V(32 x dh): A-frag from LDS (b128), B from V^T --
    const v8bf* pl = (const v8bf*)(&Plds[r * 32]);
    const v16bf pa = cat16(pl[g], pl[2 + g]);
#pragma unroll
    for (int c = 0; c < 4; ++c) {
      const v8bf* vv = (const v8bf*)(Vt + ((size_t)(h * DHd + c * 16 + r)) * KVLEN + kt);
      acc[c] = WMMA_BF16(pa, cat16(vv[g], vv[2 + g]), acc[c]);
    }
    __syncthreads();                       // before Plds is overwritten
  }

  // ---- epilogue: 1/l normalize, sigmoid gate, bf16 merged-head store ----
#pragma unroll
  for (int i = 0; i < 8; ++i) {
    const int qi = q0 + g * 8 + i;
    const float gate = 1.f / (1.f + __expf(-Gp[qi * Hh + h]));
    const float sc = gate / lrow[i];
#pragma unroll
    for (int c = 0; c < 4; ++c)
      Out[(size_t)qi * Dm + h * DHd + c * 16 + r] = (__bf16)(acc[c][i] * sc);
  }
}

// ---------------------------------------------------------------------------
// Host-side launcher.
// Inputs: x, cache_k, cache_v, episode_ids(int32), Wq, Wkv, Wg, Wo.
// Output: fp32 [2048, 512].
// ---------------------------------------------------------------------------
extern "C" void kernel_launch(void* const* d_in, const int* in_sizes, int n_in,
                              void* d_out, int out_size, void* d_ws, size_t ws_size,
                              hipStream_t stream) {
  const float* x       = (const float*)d_in[0];
  const float* cache_k = (const float*)d_in[1];
  const float* cache_v = (const float*)d_in[2];
  const int*   eids    = (const int*)  d_in[3];
  const float* Wq      = (const float*)d_in[4];
  const float* Wkv     = (const float*)d_in[5];
  const float* Wg      = (const float*)d_in[6];
  const float* Wo      = (const float*)d_in[7];
  float* out = (float*)d_out;
  (void)ws_size; (void)in_sizes; (void)n_in; (void)out_size;

  // ---- workspace carve-up: fp32 region first, then bf16 region ----------
  float* Qp  = (float*)d_ws;                     // [2048,512]  Q projection
  float* KVp = Qp  + (size_t)Nq * Dm;            // [2048,1024] KV projection
  float* Gp  = KVp + (size_t)Nq * 2 * Dm;        // [2048,8]    gate logits

  __bf16* xbf  = (__bf16*)(Gp + (size_t)Nq * Hh);
  __bf16* WqT  = xbf  + (size_t)Nq * Dm;         // [512,512]   Wq^T
  __bf16* WkvT = WqT  + (size_t)Dm * Dm;         // [1024,512]  Wkv^T
  __bf16* WgT  = WkvT + (size_t)2 * Dm * Dm;     // [8,512]     Wg^T
  __bf16* WoT  = WgT  + (size_t)Hh * Dm;         // [512,512]   Wo^T
  __bf16* Qrb  = WoT  + (size_t)Dm * Dm;         // [2048,512]  RoPE'd Q
  __bf16* Kfb  = Qrb  + (size_t)Nq * Dm;         // [8,2560,64] K (RoPE'd)
  __bf16* Vtb  = Kfb  + (size_t)Hh * KVLEN * DHd;// [8,64,2560] V transposed
  __bf16* AObf = Vtb  + (size_t)Hh * KVLEN * DHd;// [2048,512]  attn out

  auto b256 = [](int n) { return (n + 255) / 256; };
  auto blocks4 = [](int tiles) { return (tiles + 3) / 4; };  // 4 waves/block

  // ---- one-shot bf16 conversion / transposition --------------------------
  cvt_bf16  <<<b256(Nq * Dm),       256, 0, stream>>>(x,   xbf, Nq * Dm);
  cvt_bf16_t<<<b256(Dm * Dm),       256, 0, stream>>>(Wq,  WqT,  Dm, Dm);
  cvt_bf16_t<<<b256(Dm * 2 * Dm),   256, 0, stream>>>(Wkv, WkvT, Dm, 2 * Dm);
  cvt_bf16_t<<<b256(Dm * Hh),       256, 0, stream>>>(Wg,  WgT,  Dm, Hh);
  cvt_bf16_t<<<b256(Dm * Dm),       256, 0, stream>>>(Wo,  WoT,  Dm, Dm);

  // ---- projections --------------------------------------------------------
  gemm_bf16_wmma<<<blocks4((Nq/16) * (Dm/16)),   128, 0, stream>>>(xbf, WqT,  Qp,  Nq, Dm, Dm);
  gemm_bf16_wmma<<<blocks4((Nq/16) * (2*Dm/16)), 128, 0, stream>>>(xbf, WkvT, KVp, Nq, Dm, 2*Dm);
  gemm_bf16_wmma<<<blocks4((Nq/16) * 1),         128, 0, stream>>>(xbf, WgT,  Gp,  Nq, Dm, Hh);

  // ---- RoPE + KV cache assembly (bf16 outputs) ----------------------------
  rope_scale_q <<<b256(Nq * Dm),          256, 0, stream>>>(Qp, Qrb);
  build_kv_rope<<<b256(Hh * KVLEN * DHd), 256, 0, stream>>>(cache_k, cache_v, KVp, Kfb, Vtb);

  // ---- masked flash attention + gating ------------------------------------
  attn_wmma<<<dim3(Nq / 16, Hh), 32, 0, stream>>>(Qrb, Kfb, Vtb, eids, Gp, AObf);

  // ---- output projection --------------------------------------------------
  gemm_bf16_wmma<<<blocks4((Nq/16) * (Dm/16)), 128, 0, stream>>>(AObf, WoT, out, Nq, Dm, Dm);
}
